// TwoStagePointerReasoning_79456894976504
// MI455X (gfx1250) — matrix operand order repaired and checked by reading, not compiled, not run on hardware
//
#include <hip/hip_runtime.h>
#include <hip/hip_bf16.h>

// ---------------------------------------------------------------------------
// Two-stage pointer network, restructured for MI455X (gfx1250):
//  - teacher forcing => GRU sib chain precomputed, scan parallelized over T
//  - bilinear hoisted: M[b] = parent[b] . bil_W (memory bound, VALU)
//  - emb @ Wk fused with tanh-score pass, done with v_wmma_f32_16x16x32_bf16
//  - logsumexp-at-gold instead of full softmax
// ---------------------------------------------------------------------------

#define Bsz 64
#define Lsz 2048
#define Dsz 256

typedef __attribute__((ext_vector_type(16))) __bf16 v16bf;
typedef __attribute__((ext_vector_type(8)))  __bf16 v8bf;
typedef __attribute__((ext_vector_type(8)))  float  v8f;

__device__ __forceinline__ unsigned short f32_to_bf16(float f) {
  unsigned int u = __float_as_uint(f);
  unsigned int r = u + 0x7FFFu + ((u >> 16) & 1u);   // round-to-nearest-even
  return (unsigned short)(r >> 16);
}

__device__ __forceinline__ float fast_tanh(float x) {
#if __has_builtin(__builtin_amdgcn_tanhf)
  return __builtin_amdgcn_tanhf(x);          // v_tanh_f32 on gfx1250
#elif __has_builtin(__builtin_amdgcn_tanh_f32)
  return __builtin_amdgcn_tanh_f32(x);
#else
  // Pin the gfx1250 transcendental directly; v_nop covers the TRANS-op
  // result hazard (ISA 7.4) since the compiler can't see through asm.
  float r;
  asm("v_tanh_f32 %0, %1\n\tv_nop" : "=v"(r) : "v"(x));
  return r;
#endif
}

__device__ __forceinline__ float sigmoidf(float x) {
  return 1.0f / (1.0f + __expf(-x));
}

// --------------------------- fp32 -> bf16 convert ---------------------------
__global__ void k_cvt_bf16(const float* __restrict__ in,
                           unsigned short* __restrict__ out, int n) {
  int i = (blockIdx.x * blockDim.x + threadIdx.x) * 4;
  if (i + 3 < n) {
    float4 f = *(const float4*)(in + i);
    unsigned long long p =
        (unsigned long long)f32_to_bf16(f.x)
      | ((unsigned long long)f32_to_bf16(f.y) << 16)
      | ((unsigned long long)f32_to_bf16(f.z) << 32)
      | ((unsigned long long)f32_to_bf16(f.w) << 48);
    *(unsigned long long*)(out + i) = p;
  }
}

// ----------------- Wk (e,d) -> bf16 WkT (d,e)  (WMMA B operand) -------------
__global__ void k_prep_wkT(const float* __restrict__ Wk,
                           unsigned short* __restrict__ WkT) {
  int d = blockIdx.x, e = threadIdx.x;
  WkT[d * Dsz + e] = f32_to_bf16(Wk[e * Dsz + d]);
}

// ------------------- GRU chain: S[t][b][d] = sib before step t --------------
__global__ void k_gru(const float* __restrict__ emb, const int* __restrict__ gold,
                      const float* __restrict__ query,
                      const float* __restrict__ Wih, const float* __restrict__ Whh,
                      const float* __restrict__ bih, const float* __restrict__ bhh,
                      float* __restrict__ S, int T) {
  int b = blockIdx.x, d = threadIdx.x;
  __shared__ __align__(16) float xs[Dsz];
  __shared__ __align__(16) float hs[Dsz];
  hs[d] = query[d];
  __syncthreads();
  for (int t = 0; t < T; ++t) {
    S[((size_t)t * Bsz + b) * Dsz + d] = hs[d];
    int g = gold[b * T + t];
    xs[d] = emb[((size_t)b * Lsz + g) * Dsz + d];
    __syncthreads();
    float gi[3] = {bih[d], bih[Dsz + d], bih[2 * Dsz + d]};
    float gh[3] = {bhh[d], bhh[Dsz + d], bhh[2 * Dsz + d]};
    const float4* xr = (const float4*)xs;
    const float4* hr = (const float4*)hs;
    for (int m = 0; m < 3; ++m) {
      const float4* wi = (const float4*)(Wih + ((size_t)m * Dsz + d) * Dsz);
      const float4* wh = (const float4*)(Whh + ((size_t)m * Dsz + d) * Dsz);
      float a0 = 0.f, a1 = 0.f;
      #pragma unroll 8
      for (int k = 0; k < Dsz / 4; ++k) {
        float4 xv = xr[k], wv = wi[k];
        a0 += xv.x * wv.x + xv.y * wv.y + xv.z * wv.z + xv.w * wv.w;
        float4 hv = hr[k], wv2 = wh[k];
        a1 += hv.x * wv2.x + hv.y * wv2.y + hv.z * wv2.z + hv.w * wv2.w;
      }
      gi[m] += a0; gh[m] += a1;
    }
    __syncthreads();                   // all reads of hs done
    float r = sigmoidf(gi[0] + gh[0]);
    float z = sigmoidf(gi[1] + gh[1]);
    float nn = fast_tanh(gi[2] + r * gh[2]);
    float hn = (1.f - z) * nn + z * hs[d];
    __syncthreads();
    hs[d] = hn;
    __syncthreads();
  }
}

// ---------------- M[b][k][e] = sum_d parent[b][d] * W[k][d][e] --------------
__global__ void k_bilinear(const float* __restrict__ feat,
                           const float* __restrict__ W,
                           float* __restrict__ M) {
  int k = blockIdx.x, bh = blockIdx.y, e = threadIdx.x;
  __shared__ float p[32][Dsz];
  int b0 = bh * 32;
  for (int i = threadIdx.x; i < 32 * Dsz; i += 256) {
    int bb = i >> 8, dd = i & 255;
    p[bb][dd] = feat[((size_t)(b0 + bb) * Lsz + (Lsz - 2)) * Dsz + dd];
  }
  __syncthreads();
  float acc[32];
  #pragma unroll
  for (int i = 0; i < 32; ++i) acc[i] = 0.f;
  const float* Wk_ = W + (size_t)k * Dsz * Dsz;
  for (int dd = 0; dd < Dsz; ++dd) {
    if ((dd & 31) == 0 && dd + 32 < Dsz)
      __builtin_prefetch(Wk_ + (size_t)(dd + 32) * Dsz + e, 0, 1); // global_prefetch_b8
    float w = Wk_[(size_t)dd * Dsz + e];
    #pragma unroll
    for (int bb = 0; bb < 32; ++bb) acc[bb] += p[bb][dd] * w;
  }
  #pragma unroll
  for (int bb = 0; bb < 32; ++bb)
    M[(((size_t)(b0 + bb)) * Dsz + k) * Dsz + e] = acc[bb];
}

// ----------- Q[b][t][k] = bil_b[k] + sum_e M[b][k][e] * S[t][b][e] ----------
__global__ void k_q(const float* __restrict__ M, const float* __restrict__ S,
                    const float* __restrict__ bil_b, float* __restrict__ Q, int T) {
  int t = blockIdx.x, b = blockIdx.y, k = threadIdx.x;
  __shared__ __align__(16) float s[Dsz];
  s[k] = S[((size_t)t * Bsz + b) * Dsz + k];
  __syncthreads();
  const float4* m = (const float4*)(M + ((size_t)b * Dsz + k) * Dsz);
  const float4* sv = (const float4*)s;
  float acc = bil_b[k];
  #pragma unroll 8
  for (int e = 0; e < Dsz / 4; ++e) {
    float4 mv = m[e], ss = sv[e];
    acc += mv.x * ss.x + mv.y * ss.y + mv.z * ss.z + mv.w * ss.w;
  }
  Q[((size_t)b * T + t) * Dsz + k] = acc;
}

// -------------- U[bt][d] = attn_b[d] + sum_k Q[bt][k] * Wq[k][d] ------------
__global__ void k_u(const float* __restrict__ Q, const float* __restrict__ Wq,
                    const float* __restrict__ attn_b, float* __restrict__ U) {
  int bt = blockIdx.x, d = threadIdx.x;
  __shared__ float q[Dsz];
  q[d] = Q[(size_t)bt * Dsz + d];
  __syncthreads();
  float acc = attn_b[d];
  for (int k = 0; k < Dsz; ++k) acc += q[k] * Wq[(size_t)k * Dsz + d];
  U[(size_t)bt * Dsz + d] = acc;
}

// ---- fused: pk = emb_tile @ Wk (WMMA bf16), score[t] = sum tanh(pk+u).v ----
template <int T>
__global__ __launch_bounds__(256) void k_score(
    const unsigned short* __restrict__ embb,  // bf16 emb, (B,L,D) row-major
    const unsigned short* __restrict__ wkT,   // bf16 Wk transposed: (d, e)
    const float* __restrict__ U,              // (B,T,D)
    const float* __restrict__ attn_v,
    const int* __restrict__ mask,
    float* __restrict__ scores) {             // (B,T,L), masked
  int lt = blockIdx.x, b = blockIdx.y;
  int tid = threadIdx.x, lane = tid & 31, wv = tid >> 5;
  int half = (lane < 16) ? 0 : 1;
  int l15 = lane & 15;

  __shared__ __align__(16) __bf16 aebf[16][264]; // emb tile (pad => no conflicts)
  __shared__ __align__(16) float  pk[16][260];   // proj_k tile (stride 260: b128-able)
  __shared__ __align__(16) float  ul[T][260];    // u vectors   (260%64=4: lanes t -> banks 4t..4t+3)
  __shared__ __align__(16) float  vl[Dsz];

  // stage emb tile (16 rows x 256 cols of bf16)
  {
    int r = tid >> 4, c0 = (tid & 15) * 16;
    const uint4* g = (const uint4*)(embb + ((size_t)b * Lsz + lt * 16) * Dsz
                                    + (size_t)r * Dsz + c0);
    uint4* dl = (uint4*)&aebf[r][c0];
    dl[0] = g[0]; dl[1] = g[1];
  }
  for (int i = tid; i < T * Dsz; i += 256)
    ul[i >> 8][i & 255] = U[(size_t)b * T * Dsz + i];
  vl[tid] = attn_v[tid];
  __syncthreads();

  // GEMM: wave wv produces pk columns [32*wv, 32*wv+32)
  v8f ca = {}; v8f cb = {};
  int n0 = wv * 32, n1 = n0 + 16;
  for (int e0 = 0; e0 < Dsz; e0 += 32) {
    // A fragment: lane<16 rows M=lane K{e0+0..15}; lane>=16 K{e0+16..31}
    const __bf16* ap = &aebf[l15][e0 + half * 16];
    v8bf alo = *(const v8bf*)ap;
    v8bf ahi = *(const v8bf*)(ap + 8);
    v16bf a;
    #pragma unroll
    for (int i = 0; i < 8; ++i) { a[i] = alo[i]; a[i + 8] = ahi[i]; }
    // B fragments: lane holds column N, 16 consecutive K values
    v16bf b0 = *(const v16bf*)(wkT + (size_t)(n0 + l15) * Dsz + e0 + half * 16);
    v16bf b1 = *(const v16bf*)(wkT + (size_t)(n1 + l15) * Dsz + e0 + half * 16);
    ca = __builtin_amdgcn_wmma_f32_16x16x32_bf16(false, a, false, b0, (short)0, ca, false, false);
    cb = __builtin_amdgcn_wmma_f32_16x16x32_bf16(false, a, false, b1, (short)0, cb, false, false);
  }
  // scatter accumulators: element i -> row (i + 8*half), col lane&15
  #pragma unroll
  for (int i = 0; i < 8; ++i) {
    int m = i + half * 8;
    pk[m][n0 + l15] = ca[i];
    pk[m][n1 + l15] = cb[i];
  }
  __syncthreads();

  // score pass: thread -> (l = tid>>4, t = tid&15 [+16]); float4 LDS traffic
  int l = tid >> 4;
  int gl = lt * 16 + l;
  int mk = mask[(size_t)b * Lsz + gl];
  for (int t = tid & 15; t < T; t += 16) {
    float acc = 0.f;
    #pragma unroll 4
    for (int d = 0; d < Dsz; d += 4) {
      float4 pv = *(const float4*)&pk[l][d];
      float4 uv = *(const float4*)&ul[t][d];
      float4 vv = *(const float4*)&vl[d];
      acc += fast_tanh(pv.x + uv.x) * vv.x;
      acc += fast_tanh(pv.y + uv.y) * vv.y;
      acc += fast_tanh(pv.z + uv.z) * vv.z;
      acc += fast_tanh(pv.w + uv.w) * vv.w;
    }
    scores[((size_t)b * T + t) * Lsz + gl] = mk ? acc : -1.0e9f;
  }
}

// -------- per (b,t): out = exp(score[gold]-max) / sum_l exp(score-max) ------
__global__ void k_lse(const float* __restrict__ scores, const int* __restrict__ gold,
                      float* __restrict__ out, int T) {
  int t = blockIdx.x, b = blockIdx.y, tid = threadIdx.x;
  const float* row = scores + ((size_t)b * T + t) * Lsz;
  __shared__ float red[256];
  float mx = -3.4e38f;
  for (int l = tid; l < Lsz; l += 256) mx = fmaxf(mx, row[l]);
  red[tid] = mx; __syncthreads();
  for (int s = 128; s > 0; s >>= 1) {
    if (tid < s) red[tid] = fmaxf(red[tid], red[tid + s]);
    __syncthreads();
  }
  mx = red[0]; __syncthreads();
  float sm = 0.f;
  for (int l = tid; l < Lsz; l += 256) sm += __expf(row[l] - mx);
  red[tid] = sm; __syncthreads();
  for (int s = 128; s > 0; s >>= 1) {
    if (tid < s) red[tid] += red[tid + s];
    __syncthreads();
  }
  if (tid == 0) {
    int g = gold[b * T + t];
    out[b * T + t] = __expf(row[g] - mx) / red[0];
  }
}

// ---------------------------------------------------------------------------
extern "C" void kernel_launch(void* const* d_in, const int* in_sizes, int n_in,
                              void* d_out, int out_size, void* d_ws, size_t ws_size,
                              hipStream_t stream) {
  (void)in_sizes; (void)n_in; (void)out_size; (void)ws_size;
  const float* feat    = (const float*)d_in[0];
  const int*   mask    = (const int*)d_in[1];
  const int*   gold_c  = (const int*)d_in[2];
  const int*   gold_p  = (const int*)d_in[3];
  const float* query_c = (const float*)d_in[4];
  const float* query_p = (const float*)d_in[5];

  char* ws = (char*)d_ws;
  unsigned short* embb = (unsigned short*)(ws);                 // 67,108,864 B
  unsigned short* wkT  = (unsigned short*)(ws + 67108864);      //    131,072 B
  float* S       = (float*)(ws + 67239936);                     //  2,097,152 B
  float* U       = (float*)(ws + 69337088);                     //  2,097,152 B
  float* M       = (float*)(ws + 71434240);                     // 16,777,216 B
  float* Q       = (float*)(ws + 88211456);                     //  2,097,152 B
  float* scores  = (float*)(ws + 90308608);                     // 16,777,216 B
  float* out = (float*)d_out;

  k_cvt_bf16<<<32768, 256, 0, stream>>>(feat, embb, Bsz * Lsz * Dsz);

  for (int net = 0; net < 2; ++net) {
    int base = 6 + net * 10;
    const float* bil_W  = (const float*)d_in[base + 0];
    const float* bil_b  = (const float*)d_in[base + 1];
    const float* Wq     = (const float*)d_in[base + 2];
    const float* Wk     = (const float*)d_in[base + 3];
    const float* attn_b = (const float*)d_in[base + 4];
    const float* attn_v = (const float*)d_in[base + 5];
    const float* Wih    = (const float*)d_in[base + 6];
    const float* Whh    = (const float*)d_in[base + 7];
    const float* bih    = (const float*)d_in[base + 8];
    const float* bhh    = (const float*)d_in[base + 9];
    int T = (net == 0) ? 16 : 32;
    const int*   gold  = (net == 0) ? gold_c : gold_p;
    const float* query = (net == 0) ? query_c : query_p;
    float* oseg = (net == 0) ? out : (out + Bsz * 16);

    k_prep_wkT<<<Dsz, Dsz, 0, stream>>>(Wk, wkT);
    k_gru<<<Bsz, Dsz, 0, stream>>>(feat, gold, query, Wih, Whh, bih, bhh, S, T);
    k_bilinear<<<dim3(Dsz, 2), 256, 0, stream>>>(feat, bil_W, M);
    k_q<<<dim3(T, Bsz), Dsz, 0, stream>>>(M, S, bil_b, Q, T);
    k_u<<<Bsz * T, Dsz, 0, stream>>>(Q, Wq, attn_b, U);
    if (T == 16)
      k_score<16><<<dim3(Lsz / 16, Bsz), 256, 0, stream>>>(embb, wkT, U, attn_v, mask, scores);
    else
      k_score<32><<<dim3(Lsz / 16, Bsz), 256, 0, stream>>>(embb, wkT, U, attn_v, mask, scores);
    k_lse<<<dim3(T, Bsz), 256, 0, stream>>>(scores, gold, oseg, T);
  }
}
